// INRformer_77678778515727
// MI455X (gfx1250) — matrix operand order repaired and checked by reading, not compiled, not run on hardware
//
#include <hip/hip_runtime.h>

// INRformer forward for MI455X (gfx1250): bf16 WMMA for all matmuls,
// streaming-softmax attention, fused GEMM+LayerNorm epilogues, wave32 layout.

#define BT 4096
#define DM 128

typedef __attribute__((ext_vector_type(16))) __bf16 v16bf;
typedef __attribute__((ext_vector_type(8)))  __bf16 v8bf;
typedef __attribute__((ext_vector_type(8)))  float  v8f;

static __device__ __forceinline__ v8f wmma_bf16(v16bf a, v16bf b, v8f c) {
  // D = A(16x32 bf16) x B(32x16 bf16) + C(16x16 f32)
  return __builtin_amdgcn_wmma_f32_16x16x32_bf16(false, a, false, b, (short)0, c, false, false);
}

// A fragment (16x32, MxK) from row-major bf16 matrix.
// lane<16: M=lane, e0..7 -> K=k0..k0+7, e8..15 -> K=k0+16..k0+23
// lane>=16: M=lane-16, e0..7 -> K=k0+8..15, e8..15 -> K=k0+24..31
static __device__ __forceinline__ v16bf load_a_frag(const __bf16* base, int row0, int ld, int k0, int lane) {
  int m  = lane & 15;
  int hi = lane >> 4;
  const __bf16* p = base + (size_t)(row0 + m) * ld + k0 + hi * 8;
  union { v16bf v; v8bf h[2]; } u;
  u.h[0] = *reinterpret_cast<const v8bf*>(p);
  u.h[1] = *reinterpret_cast<const v8bf*>(p + 16);
  return u.v;
}

// B fragment (32x16, KxN) where B[k][n] = w[n][k], w row-major (N x K).
// lane<16: N=lane, e0..7 -> K=k0..k0+7, e8..15 -> K=k0+16..k0+23
// lane>=16: N=lane-16, e0..7 -> K=k0+8..15, e8..15 -> K=k0+24..31
static __device__ __forceinline__ v16bf load_b_frag(const __bf16* w, int n0, int ld, int k0, int lane) {
  int n    = n0 + (lane & 15);
  int koff = k0 + ((lane >> 4) << 3);
  const __bf16* p = w + (size_t)n * ld + koff;
  union { v16bf v; v8bf h[2]; } u;
  u.h[0] = *reinterpret_cast<const v8bf*>(p);
  u.h[1] = *reinterpret_cast<const v8bf*>(p + 16);
  return u.v;
}

// ---------------- coord embed + sinusoidal PE ----------------
__global__ void embed_kernel(const float* __restrict__ coords, const float* __restrict__ ce_w,
                             const float* __restrict__ ce_b, float* __restrict__ x,
                             __bf16* __restrict__ xb) {
  int b = blockIdx.x, d = threadIdx.x;
  float c0 = coords[b * 2 + 0], c1 = coords[b * 2 + 1];
  int j = d >> 1;
  float div = __powf(10000.f, (2.f * (float)j) / 128.f);
  float pe = (d & 1) ? __cosf(c1 / div) : __sinf(c0 / div);
  float v = c0 * ce_w[d * 2 + 0] + c1 * ce_w[d * 2 + 1] + ce_b[d] + pe;
  x[(size_t)b * DM + d] = v;
  xb[(size_t)b * DM + d] = (__bf16)v;
}

__global__ void cvt_bf16_kernel(const float* __restrict__ src, __bf16* __restrict__ dst, int n) {
  int i = blockIdx.x * blockDim.x + threadIdx.x;
  if (i < n) dst[i] = (__bf16)src[i];
}

__global__ void gather_kernel(const int* __restrict__ labels, const float* __restrict__ emb,
                              __bf16* __restrict__ lab) {
  int b = blockIdx.x, d = threadIdx.x;
  lab[(size_t)b * DM + d] = (__bf16)emb[(size_t)labels[b] * DM + d];
}

// ---------------- QKV projection (4096 x 384 x 128), scale folded into Q ----------------
__global__ void qkv_kernel(const __bf16* __restrict__ xb, const __bf16* __restrict__ w,
                           const float* __restrict__ bias, __bf16* __restrict__ qb,
                           __bf16* __restrict__ kb, __bf16* __restrict__ vT) {
  int lane = threadIdx.x & 31, wv = threadIdx.x >> 5;
  int tile = blockIdx.x * 4 + wv;        // 6144 tiles of 16x16
  int bt = tile / 24, nt = tile % 24;
  int row0 = bt * 16, n0 = nt * 16;
  v8f acc = {};
  for (int k0 = 0; k0 < DM; k0 += 32) {
    v16bf a = load_a_frag(xb, row0, DM, k0, lane);
    v16bf b = load_b_frag(w, n0, DM, k0, lane);
    acc = wmma_bf16(a, b, acc);
  }
  int c = n0 + (lane & 15);
  float bv = bias[c];
  int rbase = row0 + ((lane >> 4) << 3);
  const float scale = 0.17677669529663687f;  // 1/sqrt(32)
  for (int r = 0; r < 8; ++r) {
    int row = rbase + r;
    float v = acc[r] + bv;
    if (c < 128) {                 // Q: (H, B, HD), pre-scaled
      int h = c >> 5, d = c & 31;
      qb[((size_t)h * BT + row) * 32 + d] = (__bf16)(v * scale);
    } else if (c < 256) {          // K: (H, B, HD)
      int cc = c - 128, h = cc >> 5, d = cc & 31;
      kb[((size_t)h * BT + row) * 32 + d] = (__bf16)v;
    } else {                       // V transposed: (H, HD, B)
      int cc = c - 256, h = cc >> 5, d = cc & 31;
      vT[((size_t)(h * 32 + d)) * BT + row] = (__bf16)v;
    }
  }
}

// ---------------- streaming-softmax attention: 1 wave = (head, 16-query tile) ----------------
__global__ void attn_kernel(const __bf16* __restrict__ qb, const __bf16* __restrict__ kb,
                            const __bf16* __restrict__ vT, __bf16* __restrict__ attn_out) {
  __shared__ __attribute__((aligned(32))) __bf16 plds[4][16 * 32];
  int lane = threadIdx.x & 31, wv = threadIdx.x >> 5;
  int h = blockIdx.y;
  int qt = blockIdx.x * 4 + wv;
  int row0 = qt * 16;
  const __bf16* qh = qb + (size_t)h * BT * 32;
  const __bf16* kh = kb + (size_t)h * BT * 32;
  const __bf16* vh = vT + (size_t)h * 32 * BT;
  __bf16* my = plds[wv];

  v16bf qf = load_a_frag(qh, row0, 32, 0, lane);   // Q tile lives in registers
  v8f o0 = {}, o1 = {};
  float mrow[8], lrow[8];
  for (int r = 0; r < 8; ++r) { mrow[r] = -1e30f; lrow[r] = 0.f; }

  int qoff = ((lane >> 4) << 3);
  int c0 = lane & 15;

  for (int kt = 0; kt < BT; kt += 32) {
    if (kt + 32 < BT) __builtin_prefetch(kh + (size_t)(kt + 32) * 32, 0, 0);
    // S(16x32) = Q(16x32) . K_tile^T : two WMMAs (B operand = K rows, N=key)
    v8f z = {};
    v8f s0 = wmma_bf16(qf, load_b_frag(kh, kt,      32, 0, lane), z);
    v8f s1 = wmma_bf16(qf, load_b_frag(kh, kt + 16, 32, 0, lane), z);

    float ps0[8], ps1[8];
    for (int r = 0; r < 8; ++r) {
      float cm = fmaxf(s0[r], s1[r]);
      for (int msk = 8; msk >= 1; msk >>= 1) cm = fmaxf(cm, __shfl_xor(cm, msk, 32));
      float mnew  = fmaxf(mrow[r], cm);
      float alpha = __expf(mrow[r] - mnew);
      float p0 = __expf(s0[r] - mnew);
      float p1 = __expf(s1[r] - mnew);
      float rs = p0 + p1;
      for (int msk = 8; msk >= 1; msk >>= 1) rs += __shfl_xor(rs, msk, 32);
      lrow[r] = lrow[r] * alpha + rs;
      mrow[r] = mnew;
      o0[r] *= alpha; o1[r] *= alpha;
      ps0[r] = p0; ps1[r] = p1;
    }
    // transpose P from D-layout to A-layout via LDS (bf16, 16x32 per wave)
    for (int r = 0; r < 8; ++r) {
      my[(qoff + r) * 32 + c0]      = (__bf16)ps0[r];
      my[(qoff + r) * 32 + c0 + 16] = (__bf16)ps1[r];
    }
    __syncthreads();
    {
      int m = lane & 15, hi = lane >> 4;
      const __bf16* p = my + m * 32 + hi * 8;
      union { v16bf v; v8bf hh[2]; } u;
      u.hh[0] = *reinterpret_cast<const v8bf*>(p);
      u.hh[1] = *reinterpret_cast<const v8bf*>(p + 16);
      v16bf pf = u.v;
      // O(16x32) += P(16x32) . V_tile ; B operand from transposed V (contiguous keys)
      o0 = wmma_bf16(pf, load_b_frag(vh, 0,  BT, kt, lane), o0);
      o1 = wmma_bf16(pf, load_b_frag(vh, 16, BT, kt, lane), o1);
    }
    __syncthreads();
  }
  int rbase = row0 + qoff;
  for (int r = 0; r < 8; ++r) {
    float inv = 1.f / lrow[r];
    size_t row = (size_t)(rbase + r);
    attn_out[row * DM + h * 32 + c0]      = (__bf16)(o0[r] * inv);
    attn_out[row * DM + h * 32 + 16 + c0] = (__bf16)(o1[r] * inv);
  }
}

// ---------------- fused GEMM (N=128) + bias + residual + LayerNorm ----------------
// 1 wave owns 16 rows x 128 cols -> LN stats fully intra-wave. In-place on x.
__global__ void gemm_ln_kernel(const __bf16* __restrict__ ab, const __bf16* __restrict__ wb, int K,
                               const float* __restrict__ bias, const float* __restrict__ resid,
                               const float* __restrict__ g, const float* __restrict__ beta,
                               float* __restrict__ xout, __bf16* __restrict__ xbout) {
  int lane = threadIdx.x & 31, wv = threadIdx.x >> 5;
  int row0 = blockIdx.x * 64 + wv * 16;
  v8f acc[8] = {};
  for (int k0 = 0; k0 < K; k0 += 32) {
    v16bf a = load_a_frag(ab, row0, K, k0, lane);
    for (int t = 0; t < 8; ++t)
      acc[t] = wmma_bf16(a, load_b_frag(wb, t * 16, K, k0, lane), acc[t]);
  }
  int cl = lane & 15;
  int rbase = row0 + ((lane >> 4) << 3);
  for (int r = 0; r < 8; ++r) {
    size_t row = (size_t)(rbase + r);
    float vals[8];
    float s1 = 0.f, s2 = 0.f;
    for (int t = 0; t < 8; ++t) {
      int c = t * 16 + cl;
      float v = acc[t][r] + bias[c] + resid[row * DM + c];
      vals[t] = v;
      s1 += v; s2 += v * v;
    }
    for (int msk = 8; msk >= 1; msk >>= 1) {
      s1 += __shfl_xor(s1, msk, 32);
      s2 += __shfl_xor(s2, msk, 32);
    }
    float mean = s1 * (1.f / 128.f);
    float var  = s2 * (1.f / 128.f) - mean * mean;
    float rstd = rsqrtf(var + 1e-5f);
    for (int t = 0; t < 8; ++t) {
      int c = t * 16 + cl;
      float y = (vals[t] - mean) * rstd * g[c] + beta[c];
      xout[row * DM + c]  = y;
      xbout[row * DM + c] = (__bf16)y;
    }
  }
}

// ---------------- GEMM + bias (+ optional exact GeLU), bf16 out ----------------
template <bool GELU>
__global__ void gemm_act_kernel(const __bf16* __restrict__ ab, const __bf16* __restrict__ wb, int K,
                                const float* __restrict__ bias, __bf16* __restrict__ out, int N) {
  int lane = threadIdx.x & 31, wv = threadIdx.x >> 5;
  int row0 = blockIdx.x * 64 + wv * 16;
  int n0 = blockIdx.y * 64;
  v8f acc[4] = {};
  for (int k0 = 0; k0 < K; k0 += 32) {
    v16bf a = load_a_frag(ab, row0, K, k0, lane);
    for (int t = 0; t < 4; ++t)
      acc[t] = wmma_bf16(a, load_b_frag(wb, n0 + t * 16, K, k0, lane), acc[t]);
  }
  int cl = lane & 15;
  int rbase = row0 + ((lane >> 4) << 3);
  for (int t = 0; t < 4; ++t) {
    int c = n0 + t * 16 + cl;
    float bv = bias[c];
    for (int r = 0; r < 8; ++r) {
      float v = acc[t][r] + bv;
      if (GELU) v = 0.5f * v * (1.f + erff(v * 0.70710678118f));
      out[(size_t)(rbase + r) * N + c] = (__bf16)v;
    }
  }
}

// ---------------- final head: sigmoid(x @ fin_w^T + fin_b), N=1 ----------------
__global__ void final_kernel(const float* __restrict__ x, const float* __restrict__ fw,
                             const float* __restrict__ fb, float* __restrict__ out) {
  int b = blockIdx.x * blockDim.x + threadIdx.x;
  float s = fb[0];
  for (int d = 0; d < DM; ++d) s += x[(size_t)b * DM + d] * fw[d];
  out[b] = 1.f / (1.f + __expf(-s));
}

// ---------------- workspace layout (bytes) ----------------
#define OFF_X       ((size_t)0)            // f32  B x 128   (2 MB)
#define OFF_XB      ((size_t)2097152)      // bf16 B x 128   (1 MB)
#define OFF_Q       ((size_t)3145728)      // bf16 H,B,32    (1 MB)
#define OFF_K       ((size_t)4194304)      // bf16 H,B,32    (1 MB)
#define OFF_VT      ((size_t)5242880)      // bf16 H,32,B    (1 MB)
#define OFF_ATTN    ((size_t)6291456)      // bf16 B x 128   (1 MB)
#define OFF_H       ((size_t)7340032)      // bf16 B x 512   (4 MB)
#define OFF_LAB     ((size_t)11534336)     // bf16 B x 128   (1 MB)
#define OFF_CA1     ((size_t)12582912)     // bf16 B x 128   (1 MB)
#define OFF_WSAIN   ((size_t)13631488)     // 384x128 bf16
#define OFF_WSAOUT  ((size_t)13729792)     // 128x128 bf16
#define OFF_WM1W1   ((size_t)13762560)     // 512x128 bf16
#define OFF_WM1W2   ((size_t)13893632)     // 128x512 bf16
#define OFF_WWV     ((size_t)14024704)     // 128x128 bf16
#define OFF_WCAOUT  ((size_t)14057472)     // 128x128 bf16
#define OFF_WM2W1   ((size_t)14090240)     // 512x128 bf16
#define OFF_WM2W2   ((size_t)14221312)     // 128x512 bf16

extern "C" void kernel_launch(void* const* d_in, const int* in_sizes, int n_in,
                              void* d_out, int out_size, void* d_ws, size_t ws_size,
                              hipStream_t stream) {
  const float* coords   = (const float*)d_in[0];
  const int*   labels   = (const int*)d_in[1];
  const float* ce_w     = (const float*)d_in[2];
  const float* ce_b     = (const float*)d_in[3];
  const float* emb      = (const float*)d_in[4];
  const float* sa_in_w  = (const float*)d_in[5];
  const float* sa_in_b  = (const float*)d_in[6];
  const float* sa_out_w = (const float*)d_in[7];
  const float* sa_out_b = (const float*)d_in[8];
  const float* n1_g     = (const float*)d_in[9];
  const float* n1_b     = (const float*)d_in[10];
  const float* m1_w1    = (const float*)d_in[11];
  const float* m1_b1    = (const float*)d_in[12];
  const float* m1_w2    = (const float*)d_in[13];
  const float* m1_b2    = (const float*)d_in[14];
  const float* n2_g     = (const float*)d_in[15];
  const float* n2_b     = (const float*)d_in[16];
  const float* ca_in_w  = (const float*)d_in[17];
  const float* ca_in_b  = (const float*)d_in[18];
  const float* ca_out_w = (const float*)d_in[19];
  const float* ca_out_b = (const float*)d_in[20];
  const float* n3_g     = (const float*)d_in[21];
  const float* n3_b     = (const float*)d_in[22];
  const float* m2_w1    = (const float*)d_in[23];
  const float* m2_b1    = (const float*)d_in[24];
  const float* m2_w2    = (const float*)d_in[25];
  const float* m2_b2    = (const float*)d_in[26];
  const float* n4_g     = (const float*)d_in[27];
  const float* n4_b     = (const float*)d_in[28];
  const float* fin_w    = (const float*)d_in[29];
  const float* fin_b    = (const float*)d_in[30];

  char* ws = (char*)d_ws;
  float*  x     = (float*)(ws + OFF_X);
  __bf16* xb    = (__bf16*)(ws + OFF_XB);
  __bf16* qb    = (__bf16*)(ws + OFF_Q);
  __bf16* kb    = (__bf16*)(ws + OFF_K);
  __bf16* vT    = (__bf16*)(ws + OFF_VT);
  __bf16* attnb = (__bf16*)(ws + OFF_ATTN);
  __bf16* hb    = (__bf16*)(ws + OFF_H);
  __bf16* lab   = (__bf16*)(ws + OFF_LAB);
  __bf16* ca1   = (__bf16*)(ws + OFF_CA1);
  __bf16* w_sain  = (__bf16*)(ws + OFF_WSAIN);
  __bf16* w_saout = (__bf16*)(ws + OFF_WSAOUT);
  __bf16* w_m1w1  = (__bf16*)(ws + OFF_WM1W1);
  __bf16* w_m1w2  = (__bf16*)(ws + OFF_WM1W2);
  __bf16* w_wv    = (__bf16*)(ws + OFF_WWV);
  __bf16* w_caout = (__bf16*)(ws + OFF_WCAOUT);
  __bf16* w_m2w1  = (__bf16*)(ws + OFF_WM2W1);
  __bf16* w_m2w2  = (__bf16*)(ws + OFF_WM2W2);

  // weight f32 -> bf16 (tiny, stays in L2)
  cvt_bf16_kernel<<<(49152 + 255) / 256, 256, 0, stream>>>(sa_in_w, w_sain, 49152);
  cvt_bf16_kernel<<<(16384 + 255) / 256, 256, 0, stream>>>(sa_out_w, w_saout, 16384);
  cvt_bf16_kernel<<<(65536 + 255) / 256, 256, 0, stream>>>(m1_w1, w_m1w1, 65536);
  cvt_bf16_kernel<<<(65536 + 255) / 256, 256, 0, stream>>>(m1_w2, w_m1w2, 65536);
  cvt_bf16_kernel<<<(16384 + 255) / 256, 256, 0, stream>>>(ca_in_w + 2 * DM * DM, w_wv, 16384);
  cvt_bf16_kernel<<<(16384 + 255) / 256, 256, 0, stream>>>(ca_out_w, w_caout, 16384);
  cvt_bf16_kernel<<<(65536 + 255) / 256, 256, 0, stream>>>(m2_w1, w_m2w1, 65536);
  cvt_bf16_kernel<<<(65536 + 255) / 256, 256, 0, stream>>>(m2_w2, w_m2w2, 65536);

  // x = coords @ ce_w^T + ce_b + PE
  embed_kernel<<<BT, DM, 0, stream>>>(coords, ce_w, ce_b, x, xb);

  // qkv + head split (Q pre-scaled, V transposed)
  qkv_kernel<<<(BT / 16) * 24 / 4, 128, 0, stream>>>(xb, w_sain, sa_in_b, qb, kb, vT);

  // streaming attention
  attn_kernel<<<dim3(BT / 64, 4), 128, 0, stream>>>(qb, kb, vT, attnb);

  // x = LN(x + attn @ sa_out_w^T + b)
  gemm_ln_kernel<<<BT / 64, 128, 0, stream>>>(attnb, w_saout, 128, sa_out_b, x, n1_g, n1_b, x, xb);

  // MLP1: h = gelu(x @ w1^T + b1); x = LN(x + h @ w2^T + b2)
  gemm_act_kernel<true><<<dim3(BT / 64, 8), 128, 0, stream>>>(xb, w_m1w1, 128, m1_b1, hb, 512);
  gemm_ln_kernel<<<BT / 64, 128, 0, stream>>>(hb, w_m1w2, 512, m1_b2, x, n2_g, n2_b, x, xb);

  // cross-attn (degenerate): ca = (emb[labels] @ wv^T + bv) @ ca_out_w^T + b
  gather_kernel<<<BT, DM, 0, stream>>>(labels, emb, lab);
  gemm_act_kernel<false><<<dim3(BT / 64, 2), 128, 0, stream>>>(lab, w_wv, 128, ca_in_b + 256, ca1, 128);
  gemm_ln_kernel<<<BT / 64, 128, 0, stream>>>(ca1, w_caout, 128, ca_out_b, x, n3_g, n3_b, x, xb);

  // MLP2
  gemm_act_kernel<true><<<dim3(BT / 64, 8), 128, 0, stream>>>(xb, w_m2w1, 128, m2_b1, hb, 512);
  gemm_ln_kernel<<<BT / 64, 128, 0, stream>>>(hb, w_m2w2, 512, m2_b2, x, n4_g, n4_b, x, xb);

  // sigmoid head
  final_kernel<<<BT / 256, 256, 0, stream>>>(x, fin_w, fin_b, (float*)d_out);
}
